// ODEfunc_p_merge_4D_16432544875259
// MI455X (gfx1250) — matrix operand order repaired and checked by reading, not compiled, not run on hardware
//
#include <hip/hip_runtime.h>
#include <cstdint>

// ---------- types ----------
typedef __attribute__((ext_vector_type(8)))  float   v8f;
typedef __attribute__((ext_vector_type(8)))  __bf16  v8bf;
typedef __attribute__((ext_vector_type(16))) __bf16  v16bf;

static __device__ __forceinline__ __bf16 f2bf(float f) {
    uint32_t u = __builtin_bit_cast(uint32_t, f);
    uint32_t r = (u + 0x7FFFu + ((u >> 16) & 1u)) >> 16;
    return __builtin_bit_cast(__bf16, (unsigned short)r);
}
static __device__ __forceinline__ unsigned short f2bf_bits(float f) {
    uint32_t u = __builtin_bit_cast(uint32_t, f);
    return (unsigned short)((u + 0x7FFFu + ((u >> 16) & 1u)) >> 16);
}
static __device__ __forceinline__ v16bf combine(v8bf lo, v8bf hi) {
    return __builtin_shufflevector(lo, hi, 0,1,2,3,4,5,6,7,8,9,10,11,12,13,14,15);
}

// ---------- problem constants ----------
#define N0 24576
#define N1 20480
#define N2 20480
#define DIM 512
#define MID 256
#define TM 32            // rows per workgroup tile
// Row strides in bf16 elems. 1048*2 = 2096 B = 48 mod 256: the 16 half-wave row
// addresses land on 16 distinct 16B granules covering the full 64-bank period
// -> conflict-free ds_load_b128 A-fragments. 264*2 = 528 B = 16 mod 256: also free.
#define HLD 1048         // H row stride: [x(512) | ind_fea(512)] + pad
#define FLD 1048         // F row stride (1024 used)
#define MLD 264          // H2/H3 row stride

// ---------- kernel 0: t_fea and layer-1 constant c1 = fb1 + fw1[:,512:1024] @ t_fea ----------
__global__ void tfeat_kernel(const float* __restrict__ t,
                             const float* __restrict__ tw1,
                             const float* __restrict__ tb1,
                             const float* __restrict__ tw2,
                             const float* __restrict__ tb2,
                             const float* __restrict__ fw1,
                             const float* __restrict__ fb1,
                             float* __restrict__ c1) {
    __shared__ float feat[2 * DIM];
    __shared__ float tfSh[DIM];
    const float tv = t[0];
    for (int i = 0; i < 3; ++i) {
        for (int m = threadIdx.x; m < 2 * DIM; m += blockDim.x) {
            int k = m & (DIM - 1);
            float h = 100.0f * (tv * tw1[i * DIM + k] + tb1[i * DIM + k]);
            feat[m] = (m < DIM) ? __sinf(h) : __cosf(h);
        }
        __syncthreads();
        for (int j = threadIdx.x; j < DIM; j += blockDim.x) {
            float acc = tb2[i * DIM + j];
            const float* wrow = tw2 + ((size_t)i * DIM + j) * (2 * DIM);
            for (int m = 0; m < 2 * DIM; ++m) acc += feat[m] * wrow[m];
            tfSh[j] = acc;
        }
        __syncthreads();
        for (int j = threadIdx.x; j < MID; j += blockDim.x) {
            float acc = fb1[i * MID + j];
            const float* wrow = fw1 + ((size_t)i * MID + j) * (3 * DIM) + DIM;
            for (int k = 0; k < DIM; ++k) acc += tfSh[k] * wrow[k];
            c1[i * MID + j] = acc;
        }
        __syncthreads();
    }
}

// ---------- f32 -> bf16 conversion (keeps row-major [N,K]) ----------
__global__ void cvt_kernel(const float* __restrict__ src,
                           unsigned short* __restrict__ dst, int n) {
    int idx = blockIdx.x * blockDim.x + threadIdx.x;
    int stride = gridDim.x * blockDim.x;
    for (; idx < n; idx += stride) dst[idx] = f2bf_bits(src[idx]);
}

// ---------- fw1 -> bf16, dropping t_fea columns: W1p[j][k] = fw1[j][k<512 ? k : k+512] ----------
__global__ void cvt_fw1_kernel(const float* __restrict__ src,
                               unsigned short* __restrict__ dst) {
    int idx = blockIdx.x * blockDim.x + threadIdx.x;
    int stride = gridDim.x * blockDim.x;
    const int n = 3 * MID * 1024;
    for (; idx < n; idx += stride) {
        int k   = idx & 1023;
        int row = idx >> 10;                 // i*256 + j
        int sk  = (k < DIM) ? k : (k + DIM); // skip t_fea block of K
        dst[idx] = f2bf_bits(src[(size_t)row * (3 * DIM) + sk]);
    }
}

// ---------- generic WMMA GEMM stage: A (LDS, 32 x K bf16) x W^T (global [N,K] bf16) ----------
// One nT per wave, both 16-row M-tiles accumulated together: each B fragment
// (one 32-byte global load / lane) feeds TWO v_wmma, halving L2 weight traffic.
// ACT: 0 = none -> bf16 LDS, 1 = relu -> bf16 LDS, 2 = tanh -> f32 global
template <int ACT>
static __device__ __forceinline__ void gemm_stage(
    const __bf16* A, int lda, const __bf16* W, int ldk, int K, int nTiles,
    const float* bias, __bf16* OutL, int ldo, float* OutG, int ldg,
    int wave, int lane)
{
    const int g = lane >> 4, ln = lane & 15;
    for (int nT = wave; nT < nTiles; nT += 8) {
        const int jcol = nT * 16 + ln;
        v8f acc0, acc1;
#pragma unroll
        for (int v = 0; v < 8; ++v) { acc0[v] = 0.0f; acc1[v] = 0.0f; }
        const int kIters = K >> 5;
#pragma unroll 4
        for (int kt = 0; kt < kIters; ++kt) {
            const int ka = (kt << 5) + 16 * g;
            v8bf a0lo = *(const v8bf*)(A + (size_t)ln * lda + ka);
            v8bf a0hi = *(const v8bf*)(A + (size_t)ln * lda + ka + 8);
            v8bf a1lo = *(const v8bf*)(A + (size_t)(16 + ln) * lda + ka);
            v8bf a1hi = *(const v8bf*)(A + (size_t)(16 + ln) * lda + ka + 8);
            v16bf bv  = *(const v16bf*)(W + (size_t)jcol * ldk + (kt << 5) + 16 * g);
            acc0 = __builtin_amdgcn_wmma_f32_16x16x32_bf16(
                false, combine(a0lo, a0hi), false, bv, (short)0, acc0, false, false);
            acc1 = __builtin_amdgcn_wmma_f32_16x16x32_bf16(
                false, combine(a1lo, a1hi), false, bv, (short)0, acc1, false, false);
        }
        const int n = nT * 16 + ln;
        const float bval = bias[n];
#pragma unroll
        for (int v = 0; v < 8; ++v) {
            const int r0 = v + 8 * g;
            const int r1 = 16 + v + 8 * g;
            float val0 = acc0[v] + bval;
            float val1 = acc1[v] + bval;
            if (ACT == 1) {
                val0 = val0 > 0.0f ? val0 : 0.0f;
                val1 = val1 > 0.0f ? val1 : 0.0f;
            }
            if (ACT == 2) {
                OutG[(size_t)r0 * ldg + n] = tanhf(val0);
                OutG[(size_t)r1 * ldg + n] = tanhf(val1);
            } else {
                OutL[(size_t)r0 * ldo + n] = f2bf(val0);
                OutL[(size_t)r1 * ldo + n] = f2bf(val1);
            }
        }
    }
}

// ---------- main fused kernel: 32 rows per block ----------
// launch_bounds(256, 4): cap at 256 VGPRs (4 waves/SIMD) -> no extended-VGPR MSB mode,
// matches the 2-workgroups/WGP LDS occupancy (2 x ~135 KB < 320 KB).
__global__ __launch_bounds__(256, 4) void fused_mlp(
    const float* __restrict__ x,
    const float* __restrict__ u0, const float* __restrict__ u1, const float* __restrict__ u2,
    const float* __restrict__ iw1, const float* __restrict__ ib1, const float* __restrict__ ib2,
    const float* __restrict__ fb2, const float* __restrict__ fb3,
    const float* __restrict__ c1,
    const unsigned short* __restrict__ iw2b, const unsigned short* __restrict__ w1p,
    const unsigned short* __restrict__ fw2b, const unsigned short* __restrict__ fw3b,
    float* __restrict__ out)
{
    // ubuf FIRST (low LDS addresses -> 16-bit ds immediate offsets for the heaviest
    // A-fragment stream). F (32xFLD bf16) aliased with H2/H3 (disjoint lifetimes,
    // barrier-separated).
    __shared__ alignas(32) char ubuf[TM * FLD * 2];           // 67,072 B
    // H: [x(512) | ind_fea(512)] bf16, K=1024 for layer 1
    __shared__ alignas(32) __bf16 H[TM][HLD];                 // 67,072 B
    __shared__ alignas(16) float iw1Sh[DIM], ib1Sh[DIM], uSh[TM];

    __bf16 (*F)[FLD]  = reinterpret_cast<__bf16 (*)[FLD]>(ubuf);
    __bf16 (*H2)[MLD] = reinterpret_cast<__bf16 (*)[MLD]>(ubuf);
    __bf16 (*H3)[MLD] = reinterpret_cast<__bf16 (*)[MLD]>(ubuf + TM * MLD * 2);

    // block -> (branch, tile)
    int b = blockIdx.x, i, tile, rowBase;
    if (b < N0 / TM)            { i = 0; tile = b;               rowBase = 0;       }
    else if (b < (N0 + N1) / TM){ i = 1; tile = b - N0 / TM;     rowBase = N0;      }
    else                        { i = 2; tile = b - (N0+N1)/TM;  rowBase = N0 + N1; }
    const size_t gRow0 = (size_t)rowBase + (size_t)tile * TM;
    const float* uni = (i == 0) ? u0 : ((i == 1) ? u1 : u2);
    const int uBase = tile * TM;

    const int tid = threadIdx.x, lane = tid & 31, wave = tid >> 5;

    // ---- stage: scalars (pre-scaled by omega=100), x -> H[:,0:512] ----
    for (int k = tid; k < DIM; k += 256) {
        iw1Sh[k] = 100.0f * iw1[i * DIM + k];
        ib1Sh[k] = 100.0f * ib1[i * DIM + k];
    }
    if (tid < TM) uSh[tid] = uni[uBase + tid];
    for (int idx = tid; idx < TM * DIM; idx += 256) {
        int r = idx >> 9, j = idx & (DIM - 1);
        H[r][j] = f2bf(x[(gRow0 + r) * DIM + j]);
    }
    __syncthreads();   // uSh/iw1Sh ready

    // ---- Fourier features F[r][0:512]=sin, F[r][512:1024]=cos (branch-free loops) ----
    for (int idx = tid; idx < TM * DIM; idx += 256) {
        int r = idx >> 9, m = idx & (DIM - 1);
        float h = fmaf(uSh[r], iw1Sh[m], ib1Sh[m]);
        F[r][m] = f2bf(__sinf(h));
    }
    for (int idx = tid; idx < TM * DIM; idx += 256) {
        int r = idx >> 9, m = idx & (DIM - 1);
        float h = fmaf(uSh[r], iw1Sh[m], ib1Sh[m]);
        F[r][DIM + m] = f2bf(__cosf(h));
    }
    __syncthreads();

    // ---- GEMM1: ind_fea = F [32x1024] @ iw2^T -> H[:,512:1024] (+ib2) ----
    gemm_stage<0>(&F[0][0], FLD,
                  reinterpret_cast<const __bf16*>(iw2b) + (size_t)i * DIM * (2 * DIM),
                  2 * DIM, 2 * DIM, DIM / 16, ib2 + i * DIM,
                  &H[0][0] + DIM, HLD, nullptr, 0, wave, lane);
    __syncthreads();

    // ---- GEMM2: H [32x1024] @ W1p^T + c1 -> H2 [32x256], ReLU ----
    gemm_stage<1>(&H[0][0], HLD,
                  reinterpret_cast<const __bf16*>(w1p) + (size_t)i * MID * 1024,
                  1024, 1024, MID / 16, c1 + i * MID,
                  &H2[0][0], MLD, nullptr, 0, wave, lane);
    __syncthreads();

    // ---- GEMM3: H2 -> H3 [32x256], ReLU ----
    gemm_stage<1>(&H2[0][0], MLD,
                  reinterpret_cast<const __bf16*>(fw2b) + (size_t)i * MID * MID,
                  MID, MID, MID / 16, fb2 + i * MID,
                  &H3[0][0], MLD, nullptr, 0, wave, lane);
    __syncthreads();

    // ---- GEMM4: H3 -> out [32x512], tanh, f32 global ----
    gemm_stage<2>(&H3[0][0], MLD,
                  reinterpret_cast<const __bf16*>(fw3b) + (size_t)i * DIM * MID,
                  MID, MID, DIM / 16, fb3 + i * DIM,
                  nullptr, 0, out + gRow0 * DIM, DIM, wave, lane);
}

// ---------- host launcher ----------
extern "C" void kernel_launch(void* const* d_in, const int* in_sizes, int n_in,
                              void* d_out, int out_size, void* d_ws, size_t ws_size,
                              hipStream_t stream) {
    (void)in_sizes; (void)n_in; (void)out_size; (void)ws_size;
    const float* t   = (const float*)d_in[0];
    const float* x   = (const float*)d_in[1];
    const float* u0  = (const float*)d_in[2];
    const float* u1  = (const float*)d_in[3];
    const float* u2  = (const float*)d_in[4];
    const float* tw1 = (const float*)d_in[5];
    const float* tb1 = (const float*)d_in[6];
    const float* tw2 = (const float*)d_in[7];
    const float* tb2 = (const float*)d_in[8];
    const float* iw1 = (const float*)d_in[9];
    const float* ib1 = (const float*)d_in[10];
    const float* iw2 = (const float*)d_in[11];
    const float* ib2 = (const float*)d_in[12];
    const float* fw1 = (const float*)d_in[13];
    const float* fb1 = (const float*)d_in[14];
    const float* fw2 = (const float*)d_in[15];
    const float* fb2 = (const float*)d_in[16];
    const float* fw3 = (const float*)d_in[17];
    const float* fb3 = (const float*)d_in[18];
    float* out = (float*)d_out;

    // workspace layout (bytes)
    char* ws = (char*)d_ws;
    float*          c1   = (float*)ws;                                 // 3*256 f32 (3 KB)
    unsigned short* iw2b = (unsigned short*)(ws + 8192);               // 3*512*1024 bf16
    unsigned short* w1p  = (unsigned short*)(ws + 8192 + 3145728);     // 3*256*1024 bf16
    unsigned short* fw2b = (unsigned short*)(ws + 8192 + 3145728 + 1572864);
    unsigned short* fw3b = (unsigned short*)(ws + 8192 + 3145728 + 1572864 + 393216);

    tfeat_kernel<<<1, 256, 0, stream>>>(t, tw1, tb1, tw2, tb2, fw1, fb1, c1);
    cvt_kernel<<<1024, 256, 0, stream>>>(iw2, iw2b, 3 * 512 * 1024);
    cvt_fw1_kernel<<<512, 256, 0, stream>>>(fw1, w1p);
    cvt_kernel<<<256, 256, 0, stream>>>(fw2, fw2b, 3 * 256 * 256);
    cvt_kernel<<<512, 256, 0, stream>>>(fw3, fw3b, 3 * 512 * 256);

    const int nBlocks = (N0 + N1 + N2) / TM;   // 2048
    fused_mlp<<<nBlocks, 256, 0, stream>>>(
        x, u0, u1, u2, iw1, ib1, ib2, fb2, fb3,
        c1, iw2b, w1p, fw2b, fw3b, out);
}
